// SelfAttention_56564719288804
// MI455X (gfx1250) — matrix operand order repaired and checked by reading, not compiled, not run on hardware
//
#include <hip/hip_runtime.h>
#include <hip/hip_bf16.h>
#include <math.h>

// Problem constants
#define BATCH 32
#define LSEQ  512
#define DMODEL 128
#define NHEAD 8
#define DHEAD 16

typedef _Float16 half_t;
typedef __attribute__((ext_vector_type(16))) _Float16 v16h;
typedef __attribute__((ext_vector_type(8)))  _Float16 v8h;
typedef __attribute__((ext_vector_type(8)))  float    v8f;
typedef __attribute__((ext_vector_type(4)))  float    v4f;

static __device__ __forceinline__ v16h cat8(v8h a, v8h b) {
  return __builtin_shufflevector(a, b, 0,1,2,3,4,5,6,7,8,9,10,11,12,13,14,15);
}

#define WMMA_F16(A, B, C) \
  __builtin_amdgcn_wmma_f32_16x16x32_f16(false, (A), false, (B), (short)0, (C), false, false)

// ---------------------------------------------------------------------------
// Kernel 0: pack queries [b][d][l] f32 -> qT [b][l][d] f16,
//           and W_mem(256x128)+W_q(128x128) -> Wc (384x128) f16
// ---------------------------------------------------------------------------
__global__ void k_pack(const float* __restrict__ qsrc,
                       const float* __restrict__ Wm,
                       const float* __restrict__ Wq,
                       half_t* __restrict__ qT,
                       half_t* __restrict__ Wc) {
  int idx = blockIdx.x * blockDim.x + threadIdx.x;
  const int NQ = BATCH * DMODEL * LSEQ;           // 2^21
  if (idx < NQ) {
    int l = idx & (LSEQ - 1);
    int d = (idx >> 9) & (DMODEL - 1);
    int b = idx >> 16;                            // 128*512 = 2^16
    qT[((size_t)b * LSEQ + l) * DMODEL + d] = (half_t)qsrc[idx];
  } else {
    int k = idx - NQ;
    if (k < 384 * DMODEL) {
      float v = (k < 256 * DMODEL) ? Wm[k] : Wq[k - 256 * DMODEL];
      Wc[k] = (half_t)v;
    }
  }
}

// ---------------------------------------------------------------------------
// Kernel 1: yc[b][l][o] = sum_d Wc[o][d] * qT[b][l][d]   (o in 0..383)
// One wave per 16x16 output tile; K=128 via 4x wmma_f32_16x16x32_f16,
// split into two independent accumulator chains to dodge D->C RAW NOPs.
// ---------------------------------------------------------------------------
__global__ __launch_bounds__(32) void k_gemm(const half_t* __restrict__ Wc,
                                             const half_t* __restrict__ qT,
                                             float* __restrict__ yc) {
  const int wg = blockIdx.x;
  const int mt = wg % 24;              // 384/16 output-feature tiles
  const int lt = (wg / 24) % 32;       // 512/16 position tiles
  const int b  = wg / (24 * 32);
  const int lane = threadIdx.x;
  const int hi = lane >> 4;            // half-wave: 0 or 1
  const int rl = lane & 15;

  const half_t* arow = Wc + (size_t)(mt * 16 + rl) * DMODEL;            // A row (M = rl)
  const half_t* bcol = qT + ((size_t)b * LSEQ + lt * 16 + rl) * DMODEL; // B col (N = rl)

  v8f c0 = {};
  v8f c1 = {};
  for (int kc = 0; kc < 4; kc += 2) {
    // A lane layout: lower half K={0..7,16..23}+32kc, upper half K={8..15,24..31}+32kc
    int ka0 = kc * 32 + (hi ? 8 : 0);
    int ka1 = ka0 + 32;
    v16h a0 = cat8(*(const v8h*)(arow + ka0), *(const v8h*)(arow + ka0 + 16));
    v16h a1 = cat8(*(const v8h*)(arow + ka1), *(const v8h*)(arow + ka1 + 16));
    // B lane layout: lower half K={0..7,16..23}+32kc, upper half K={8..15,24..31}+32kc
    int kb0 = kc * 32 + (hi ? 8 : 0);
    v16h b0 = cat8(*(const v8h*)(bcol + kb0), *(const v8h*)(bcol + kb0 + 16));
    v16h b1 = cat8(*(const v8h*)(bcol + kb0 + 32), *(const v8h*)(bcol + kb0 + 48));
    c0 = WMMA_F16(a0, b0, c0);
    c1 = WMMA_F16(a1, b1, c1);
  }
  v8f c = c0 + c1;
  // C layout: lane holds col N=rl, rows M = e + hi*8 for e=0..7 (consecutive o)
  float* orow = yc + ((size_t)b * LSEQ + lt * 16 + rl) * 384 + mt * 16 + hi * 8;
  v4f lo = {c[0], c[1], c[2], c[3]};
  v4f hi4 = {c[4], c[5], c[6], c[7]};
  *(v4f*)orow = lo;
  *(v4f*)(orow + 4) = hi4;
}

// ---------------------------------------------------------------------------
// Kernel 2: hyperbolic projections per (b,h,l).
//   q_hp[b,h,l,32] f16 : [ q_spatial*sinh/|q| (16), cosh (1), zeros (15) ]
//   k_hp[b,h,l,32] f16 : [ -k_spatial*sinh/|k| (16), cosh (1), zeros (15) ]
//        => wmma(k_hp, q_hp) = -(k.q - kt*qt) = -sp = arccosh argument
//   v_kT[b,h,d,512] f16 : Klein coords, transposed for AV B-operand
//   lb[b,h,l] f32      : log(1/denom) + bias + mask_penalty
// ---------------------------------------------------------------------------
__global__ void k_proj(const float* __restrict__ yc,
                       const int* __restrict__ mask,
                       const float* __restrict__ bias_p,
                       half_t* __restrict__ q_hp,
                       half_t* __restrict__ k_hp,
                       half_t* __restrict__ v_kT,
                       float* __restrict__ lb) {
  int idx = blockIdx.x * blockDim.x + threadIdx.x;
  if (idx >= BATCH * NHEAD * LSEQ) return;
  int l = idx & (LSEQ - 1);
  int h = (idx >> 9) & (NHEAD - 1);
  int b = idx >> 12;
  const float* base = yc + ((size_t)b * LSEQ + l) * 384;
  const float* Kf = base + h * DHEAD;          // memory[..., :D]
  const float* Vf = base + 128 + h * DHEAD;    // memory[..., D:]
  const float* Qf = base + 256 + h * DHEAD;    // query
  size_t bh = (size_t)b * NHEAD + h;

  // --- Q -> hyperboloid ---
  float q[DHEAD]; float qn2 = 0.f;
  for (int d = 0; d < DHEAD; ++d) { q[d] = Qf[d] * 0.25f; qn2 += q[d] * q[d]; }
  float qn = sqrtf(qn2);
  float qsh = sinhf(qn), qch = coshf(qn);
  float qi = qsh / fmaxf(qn, 1e-12f);
  half_t* qo = q_hp + (bh * LSEQ + l) * 32;
  for (int d = 0; d < DHEAD; ++d) qo[d] = (half_t)(q[d] * qi);
  qo[16] = (half_t)qch;
  for (int d = 17; d < 32; ++d) qo[d] = (half_t)0.f;

  // --- K -> hyperboloid, spatial negated (fold Lorentz signature) ---
  float k[DHEAD]; float kn2 = 0.f;
  for (int d = 0; d < DHEAD; ++d) { k[d] = Kf[d]; kn2 += k[d] * k[d]; }
  float kn = sqrtf(kn2);
  float ksh = sinhf(kn), kch = coshf(kn);
  float ki = -ksh / fmaxf(kn, 1e-12f);
  half_t* ko = k_hp + (bh * LSEQ + l) * 32;
  for (int d = 0; d < DHEAD; ++d) ko[d] = (half_t)(k[d] * ki);
  ko[16] = (half_t)kch;
  for (int d = 17; d < 32; ++d) ko[d] = (half_t)0.f;

  // --- V -> Klein ---
  float v[DHEAD]; float vn2 = 0.f;
  for (int d = 0; d < DHEAD; ++d) { v[d] = Vf[d]; vn2 += v[d] * v[d]; }
  float vn = sqrtf(vn2);
  float t = tanhf(vn);                       // h_spatial/h_t = normalize(V)*tanh(|V|)
  float vi = t / fmaxf(vn, 1e-12f);
  float vk[DHEAD]; float n2 = 0.f;
  for (int d = 0; d < DHEAD; ++d) { vk[d] = v[d] * vi; n2 += vk[d] * vk[d]; }
  float nn = sqrtf(n2);
  float inv2 = 1.f / fmaxf(nn, 1e-12f);
  float vkn2 = 0.f;
  for (int d = 0; d < DHEAD; ++d) {
    vk[d] *= inv2; vkn2 += vk[d] * vk[d];
    v_kT[(bh * DHEAD + d) * LSEQ + l] = (half_t)vk[d];
  }
  float vkn = sqrtf(vkn2);
  float dm = vkn - 1e-4f;
  float denom = sqrtf(1.f - dm * dm) + 1e-4f;
  float lv = logf(1.f / denom) + bias_p[0];
  if (mask[b * LSEQ + l] == 0) lv += -1e30f;
  lb[bh * LSEQ + l] = lv;
}

// ---------------------------------------------------------------------------
// Kernel 3: one wave per (b, h, 16-row i-tile).
//  Phase A: 32 score WMMAs -> arccosh (fast: log(x+sqrt(x^2-1))) -> LDS logits
//  Phase B: row softmax (shfl_xor butterflies), weights -> f16 LDS
//  Phase C: 16 accumulate WMMAs (W x V) in two chains, store channels-first
// ---------------------------------------------------------------------------
__global__ __launch_bounds__(32) void k_attn(const half_t* __restrict__ q_hp,
                                             const half_t* __restrict__ k_hp,
                                             const half_t* __restrict__ v_kT,
                                             const float* __restrict__ lb,
                                             const float* __restrict__ beta_p,
                                             float* __restrict__ out) {
  __shared__ float  sl[16][LSEQ];   // logits (32 KB)
  __shared__ half_t sw[16][LSEQ];   // softmax weights f16 (16 KB)

  int wg = blockIdx.x;
  int it = wg & 31;                 // i tile (rows of K side)
  int h  = (wg >> 5) & 7;
  int b  = wg >> 8;
  int lane = threadIdx.x;
  int hi = lane >> 4;
  int rl = lane & 15;
  size_t bh = (size_t)b * NHEAD + h;
  float beta = beta_p[0];

  // A operand: k_hp rows i = it*16 + rl  (lane layout per ISA 16-bit A 16x32)
  const half_t* krow = k_hp + (bh * LSEQ + it * 16 + rl) * 32;
  int ka = hi ? 8 : 0;
  v16h aK = cat8(*(const v8h*)(krow + ka), *(const v8h*)(krow + ka + 16));

  const float* lbrow = lb + bh * LSEQ;

  for (int jt = 0; jt < 32; ++jt) {
    int j = jt * 16 + rl;
    const half_t* qcol = q_hp + (bh * LSEQ + j) * 32 + (hi ? 8 : 0);
    v16h bm = cat8(*(const v8h*)qcol, *(const v8h*)(qcol + 16));
    v8f c = {};
    c = WMMA_F16(aK, bm, c);                  // c = -sp  (>= 1 on hyperboloid)
    float lbj = lbrow[j];
    for (int e = 0; e < 8; ++e) {
      float x = c[e] + 1e-6f;
      // arccosh(x) = log(x + sqrt(x*x - 1))  (== ref's log(x)+log1p(sqrt/x))
      float dist = __logf(x + sqrtf(x * x - 1.f));
      sl[e + hi * 8][j] = -beta * dist + lbj;
    }
  }
  __syncthreads();

  // Row-wise softmax over 512 columns; 32 lanes strided by 32.
  for (int r = 0; r < 16; ++r) {
    float m = -3.0e38f;
    for (int t = 0; t < 16; ++t) m = fmaxf(m, sl[r][lane + 32 * t]);
    for (int off = 16; off >= 1; off >>= 1) m = fmaxf(m, __shfl_xor(m, off, 32));
    float s = 0.f;
    for (int t = 0; t < 16; ++t) {
      float e = __expf(sl[r][lane + 32 * t] - m);
      sl[r][lane + 32 * t] = e;
      s += e;
    }
    for (int off = 16; off >= 1; off >>= 1) s += __shfl_xor(s, off, 32);
    float inv = 1.f / s;
    for (int t = 0; t < 16; ++t)
      sw[r][lane + 32 * t] = (half_t)(sl[r][lane + 32 * t] * inv);
  }
  __syncthreads();

  // AV: out[i,d] = sum_j w[i,j] * v_k[j,d];  M=i, N=d, K=j (512 = 16 x 32)
  // Two independent accumulator chains to avoid WMMA->WMMA RAW NOPs.
  v8f acc0 = {};
  v8f acc1 = {};
  const half_t* vbase = v_kT + (bh * DHEAD + rl) * LSEQ + (hi ? 8 : 0);
  for (int jt = 0; jt < 16; jt += 2) {
    int kw0 = jt * 32 + (hi ? 8 : 0);
    int kw1 = kw0 + 32;
    v16h aw0 = cat8(*(const v8h*)&sw[rl][kw0], *(const v8h*)&sw[rl][kw0 + 16]);
    v16h aw1 = cat8(*(const v8h*)&sw[rl][kw1], *(const v8h*)&sw[rl][kw1 + 16]);
    v16h bv0 = cat8(*(const v8h*)(vbase + jt * 32), *(const v8h*)(vbase + jt * 32 + 16));
    v16h bv1 = cat8(*(const v8h*)(vbase + jt * 32 + 32), *(const v8h*)(vbase + jt * 32 + 48));
    acc0 = WMMA_F16(aw0, bv0, acc0);
    acc1 = WMMA_F16(aw1, bv1, acc1);
  }
  v8f acc = acc0 + acc1;
  // out[b][h*16 + d][l],  d = rl (lane col),  l = it*16 + hi*8 + e  (contiguous)
  float* op = out + ((size_t)b * DMODEL + h * DHEAD + rl) * LSEQ + it * 16 + hi * 8;
  v4f lo = {acc[0], acc[1], acc[2], acc[3]};
  v4f hi4 = {acc[4], acc[5], acc[6], acc[7]};
  *(v4f*)op = lo;
  *(v4f*)(op + 4) = hi4;
}

// ---------------------------------------------------------------------------
extern "C" void kernel_launch(void* const* d_in, const int* in_sizes, int n_in,
                              void* d_out, int out_size, void* d_ws, size_t ws_size,
                              hipStream_t stream) {
  const float* queries = (const float*)d_in[0];   // [B, D, L]
  const int*   mask    = (const int*)d_in[1];     // [B, L]
  const float* W_mem   = (const float*)d_in[2];   // [256, 128]
  const float* W_q     = (const float*)d_in[3];   // [128, 128]
  const float* beta    = (const float*)d_in[4];   // [1]
  const float* bias    = (const float*)d_in[5];   // [1]
  float* out = (float*)d_out;                     // [B, D, L]

  // Workspace layout (256B aligned sub-buffers)
  char* ws = (char*)d_ws;
  size_t off = 0;
  auto take = [&](size_t bytes) -> char* {
    char* p = ws + off;
    off = (off + bytes + 255) & ~(size_t)255;
    return p;
  };
  half_t* qT   = (half_t*)take((size_t)BATCH * LSEQ * DMODEL * 2);        // 4 MB
  half_t* Wc   = (half_t*)take((size_t)384 * DMODEL * 2);                 // 96 KB
  float*  yc   = (float*) take((size_t)BATCH * LSEQ * 384 * 4);           // 24 MB
  half_t* q_hp = (half_t*)take((size_t)BATCH * NHEAD * LSEQ * 32 * 2);    // 8 MB
  half_t* k_hp = (half_t*)take((size_t)BATCH * NHEAD * LSEQ * 32 * 2);    // 8 MB
  half_t* v_kT = (half_t*)take((size_t)BATCH * NHEAD * DHEAD * LSEQ * 2); // 4 MB
  float*  lbuf = (float*) take((size_t)BATCH * NHEAD * LSEQ * 4);         // 512 KB
  (void)ws_size; (void)in_sizes; (void)n_in; (void)out_size;

  // K0: pack/transpose inputs to f16 WMMA-friendly layouts
  {
    int n = BATCH * DMODEL * LSEQ + 384 * DMODEL;
    k_pack<<<(n + 255) / 256, 256, 0, stream>>>(queries, W_mem, W_q, qT, Wc);
  }
  // K1: fused projection GEMM -> yc [B, L, 384]
  {
    int blocks = BATCH * 32 * 24;   // (b, l-tile, o-tile)
    k_gemm<<<blocks, 32, 0, stream>>>(Wc, qT, yc);
  }
  // K2: hyperbolic/Klein projections + per-position logit bias
  {
    int n = BATCH * NHEAD * LSEQ;
    k_proj<<<(n + 255) / 256, 256, 0, stream>>>(yc, mask, bias, q_hp, k_hp, v_kT, lbuf);
  }
  // K3: Lorentz attention (scores WMMA -> arccosh -> softmax -> AV WMMA)
  {
    int blocks = BATCH * NHEAD * (LSEQ / 16);   // 8192 waves
    k_attn<<<blocks, 32, 0, stream>>>(q_hp, k_hp, v_kT, lbuf, beta, out);
  }
}
